// AggressiveForgettingController_55087250538935
// MI455X (gfx1250) — matrix operand order repaired and checked by reading, not compiled, not run on hardware
//
#include <hip/hip_runtime.h>
#include <hip/hip_bf16.h>
#include <math.h>

typedef __attribute__((ext_vector_type(16))) _Float16 v16h;
typedef __attribute__((ext_vector_type(8)))  _Float16 v8h;
typedef __attribute__((ext_vector_type(8)))  float    v8f;
typedef unsigned long long ull;

constexpr int NB = 32, ND = 512, NH = 256, NS = 131072;
constexpr float CAP_LIMIT = 0.85f, MIN_NOV = 0.4f, STORE_TH = 0.35f, ACT_TH = 0.5f, STAB = 0.1f;

// ---- d_out layout (flat, return order) ----
constexpr size_t OUT_MEM = 0;
constexpr size_t OUT_AT  = (size_t)NS * ND;        // 67108864
constexpr size_t OUT_ER  = OUT_AT + NS;            // 67239936
constexpr size_t OUT_SC  = OUT_ER + NS;            // 67371008 (7 scalars)

// ---- workspace float-element offsets ----
constexpr int WS_CAP=0, WS_AGEMAX=1, WS_VICT=2 /*ull*/, WS_WPACK=4 /*ull*/, WS_CONFCNT=6,
  WS_OVERCAP=7, WS_VICTIDX=8, WS_WIDX=9, WS_SHOULD=10, WS_STMEAN=11, WS_NOVMEAN=12,
  WS_RELEV=13, WS_LISTCNT=15, WS_CAPF=16,
  WS_LIST=32,          // 128 ints
  WS_MAXSIM=160,       // 32 f
  WS_TOPV=192,         // 96 f
  WS_TOPI=288,         // 96 int
  WS_STR=384,          // 96 f
  WS_ACT=480,          // 96 int
  WS_WIN=576,          // 96 int
  WS_SSC=672,          // 32 f
  WS_DRIFT0=704,       // 512 f
  WS_ROWSUM=1216,      // 128 f
  WS_QN=1408,          // 32*512 halfs = 8192 f
  WS_CW=9600,          // 8192 f
  WS_NORMS=17792,      // NS f
  WS_H1=148864,        // 32*256
  WS_HLN=157056,       // 32*256
  WS_NOVLIN=165248,    // 32*128
  WS_HD=169344,        // 96*256
  WS_HS=193920,        // 96*256
  WS_OLD=218496,       // 96*512
  WS_SIMS=267648;      // 32*NS floats (~16MB); total ws use ~17.9MB

__device__ __forceinline__ float sigf(float x){ return 1.f/(1.f+expf(-x)); }
__device__ __forceinline__ unsigned ordkey(float f){
  unsigned b=__float_as_uint(f); return (b&0x80000000u)? ~b : (b|0x80000000u);
}
__device__ __forceinline__ ull shfl_xor_u64(ull v, int m){
  unsigned lo=(unsigned)v, hi=(unsigned)(v>>32);
  lo=__shfl_xor(lo,m); hi=__shfl_xor(hi,m);
  return ((ull)hi<<32)|(ull)lo;
}

// ---------- K0: init scalars, build f16 normalized queries + f16 conf_w ----------
__global__ void k0_prep(const float* nc, const float* conf_w, float* W){
  __shared__ float red[256];
  int b = blockIdx.x, t = threadIdx.x;
  if (b==0 && t==0){
    ((unsigned*)W)[WS_CAP]=0u; ((unsigned*)W)[WS_AGEMAX]=0u;
    *(ull*)(W+WS_VICT)=0ull;   *(ull*)(W+WS_WPACK)=0ull;
    ((unsigned*)W)[WS_CONFCNT]=0u;
  }
  float p=0.f;
  for(int k=t;k<ND;k+=256){ float x=nc[b*ND+k]; p+=x*x; }
  red[t]=p; __syncthreads();
  for(int o=128;o>0;o>>=1){ if(t<o) red[t]+=red[t+o]; __syncthreads(); }
  float invn = 1.f / fmaxf(sqrtf(red[0]), 1e-12f);
  _Float16* Qh=(_Float16*)(W+WS_QN); _Float16* Ch=(_Float16*)(W+WS_CW);
  for(int k=t;k<ND;k+=256){
    Qh[b*ND+k]=(_Float16)(nc[b*ND+k]*invn);
    Ch[b*ND+k]=(_Float16)(conf_w[b*ND+k]);
  }
}

// ---------- K1: age.max() ----------
__global__ void k1_agemax(const float* at, const int* stepp, float* W){
  int s = blockIdx.x*blockDim.x + threadIdx.x;
  float step=(float)stepp[0];
  float age=fmaxf(step-at[s],0.f);             // non-negative -> bit-order == value-order
  atomicMax((unsigned*)W + WS_AGEMAX, __float_as_uint(age));
}

// ---------- K2: fused big pass over memory bank (WMMA) ----------
// Per wave: 16 rows. Pass A: copy row to out, row norms, capacity, victim argmax, LDS f16 stage.
// Pass B: WMMA sims (Qn x memT, post-scaled by invnorm) and conf->erase_scores (mem x conf_wT).
__global__ __launch_bounds__(256) void k2_main(const float* mem0, const float* at_in,
    float* mem_out, float* at_out, float* erase_out, float* W,
    const float* conf_b, const float* eg_w, const float* eg_b,
    const float* lru_w, const float* lru_b, const int* stepp)
{
  extern __shared__ char dynsmem[];
  const int lane = threadIdx.x & 31, wv = threadIdx.x >> 5;
  const int s0 = (blockIdx.x*8 + wv)*16;
  _Float16* tile = (_Float16*)dynsmem + wv*(16*ND);
  const float step = (float)stepp[0];

  // ---- pass A: 2 lanes per row, 256 floats each ----
  const int row = lane>>1, half = lane&1;
  const int sr = s0 + row;
  const float* src = mem0 + (size_t)sr*ND + half*256;
  float*       dst = mem_out + (size_t)sr*ND + half*256;
  _Float16*    tp  = tile + row*ND + half*256;
  float ss=0.f;
  for(int i=0;i<256;i+=4){
    float4 v = *(const float4*)(src+i);
    *(float4*)(dst+i) = v;
    ss += v.x*v.x+v.y*v.y+v.z*v.z+v.w*v.w;
    tp[i]=(_Float16)v.x; tp[i+1]=(_Float16)v.y; tp[i+2]=(_Float16)v.z; tp[i+3]=(_Float16)v.w;
  }
  ss += __shfl_xor(ss,1);
  float nrm  = sqrtf(ss);
  float invn = 1.f/fmaxf(nrm,1e-12f);
  ull key = 0ull;
  if(half==0){
    W[WS_NORMS+sr]=nrm;
    at_out[sr]=at_in[sr];
    float agemax = __uint_as_float(((const unsigned*)W)[WS_AGEMAX]);
    float age = fmaxf(step-at_in[sr],0.f);
    float esc = age/(agemax+1e-6f) + (1.f - sigf(nrm));
    key = (((ull)ordkey(esc))<<32) | (ull)(0xFFFFFFFFu-(unsigned)sr); // first-index tiebreak
  }
  for(int o=16;o>0;o>>=1){ ull oth=shfl_xor_u64(key,o); if(oth>key) key=oth; }
  if(lane==0) atomicMax((ull*)(W+WS_VICT), key);
  ull bal = __ballot(half==0 && nrm>ACT_TH);
  if(lane==0) atomicAdd((unsigned*)W+WS_CAP, (unsigned)__popcll(bal));
  __syncthreads();

  // ---- pass B: WMMA ----
  const int hi=lane>>4, col=lane&15;
  float invn_col = __shfl(invn, col*2);   // invnorm of s-row 'col' in this tile
  v8f aS0={},aS1={},aC0={},aC1={};
  const _Float16* Qh=(const _Float16*)(W+WS_QN);
  const _Float16* Ch=(const _Float16*)(W+WS_CW);
  const _Float16* trow = tile + col*ND;
  for(int kc=0;kc<16;++kc){
    int kb=kc*32;
    v16h fB, fAm, fQ0, fQ1, fW0, fW1;
    // B (sims): element e -> K = kb + e + 8*hi (+16 for e>=8), column = col (mem row 'col')
    { v8h b0 = *(const v8h*)(trow+kb+8*hi);
      v8h b1 = *(const v8h*)(trow+kb+8*hi+16);
      for(int e=0;e<8;++e){ fB[e]=b0[e]; fB[8+e]=b1[e]; } }
    // A (conf): row = col, K_A(e,hi) = e + 16*hi
    { v8h a0 = *(const v8h*)(trow+kb+16*hi);
      v8h a1 = *(const v8h*)(trow+kb+16*hi+8);
      for(int e=0;e<8;++e){ fAm[e]=a0[e]; fAm[8+e]=a1[e]; } }
    // A (sims) = Qn rows (two 16-row b-tiles)
    { const _Float16* q0 = Qh + (size_t)col*ND      + kb + 8*hi;
      const _Float16* q1 = Qh + (size_t)(16+col)*ND + kb + 8*hi;
      v8h x0=*(const v8h*)q0, x1=*(const v8h*)(q0+16);
      v8h y0=*(const v8h*)q1, y1=*(const v8h*)(q1+16);
      for(int e=0;e<8;++e){ fQ0[e]=x0[e]; fQ0[8+e]=x1[e]; fQ1[e]=y0[e]; fQ1[8+e]=y1[e]; } }
    // B (conf) = conf_w rows as columns (two 16-col j-tiles)
    { const _Float16* w0 = Ch + (size_t)col*ND      + kb + 16*hi;
      const _Float16* w1 = Ch + (size_t)(16+col)*ND + kb + 16*hi;
      v8h x0=*(const v8h*)w0, x1=*(const v8h*)(w0+8);
      v8h y0=*(const v8h*)w1, y1=*(const v8h*)(w1+8);
      for(int e=0;e<8;++e){ fW0[e]=x0[e]; fW0[8+e]=x1[e]; fW1[e]=y0[e]; fW1[8+e]=y1[e]; } }
    aS0=__builtin_amdgcn_wmma_f32_16x16x32_f16(false,fQ0,false,fB,(short)0,aS0,false,false);
    aS1=__builtin_amdgcn_wmma_f32_16x16x32_f16(false,fQ1,false,fB,(short)0,aS1,false,false);
    aC0=__builtin_amdgcn_wmma_f32_16x16x32_f16(false,fAm,false,fW0,(short)0,aC0,false,false);
    aC1=__builtin_amdgcn_wmma_f32_16x16x32_f16(false,fAm,false,fW1,(short)0,aC1,false,false);
  }
  // sims out: D-layout M=b=v+8*hi, N=s=col ; post-scale by invnorm of column
  float* sims = W + WS_SIMS;
  int sc = s0 + col;
  for(int v=0;v<8;++v){
    int b0=v+8*hi;
    sims[(size_t)b0*NS + sc]      = aS0[v]*invn_col;
    sims[(size_t)(16+b0)*NS + sc] = aS1[v]*invn_col;
  }
  // conf D-layout: M = s-row = v+8*hi, N = j = col (+16 for tile1); reduce over j (16 lanes/half)
  float pm[8];
  for(int v=0;v<8;++v)
    pm[v] = eg_w[32+col]*sigf(aC0[v]+conf_b[col]) + eg_w[48+col]*sigf(aC1[v]+conf_b[16+col]);
  for(int o=1;o<16;o<<=1) for(int v=0;v<8;++v) pm[v] += __shfl_xor(pm[v],o);
  if(col<8){
    int sr2 = s0 + hi*8 + col;
    float sum = pm[col];
    float a = (step - at_in[sr2])*1e-3f;      // lru input: (step-at)/1000 (no clamp, per ref)
    for(int k=0;k<32;++k) sum += eg_w[k]*fmaxf(a*lru_w[k]+lru_b[k],0.f);
    erase_out[sr2] = sigf(sum + eg_b[0]);
  }
}

// ---------- K3: decode victim/over_cap, apply emergency erase + erase_score fixup ----------
__global__ void k3_fix(float* mem_out, float* at_out, float* erase_out, float* W,
    const float* conf_b, const float* eg_w, const float* eg_b,
    const float* lru_w, const float* lru_b, const int* stepp)
{
  __shared__ int svict, sover;
  int t=threadIdx.x;
  if(t==0){
    ull key = *(ull*)(W+WS_VICT);
    int vict = (int)(0xFFFFFFFFu - (unsigned)(key & 0xFFFFFFFFull));
    float cap = (float)(((unsigned*)W)[WS_CAP])/(float)NS;
    int over = cap > CAP_LIMIT;
    ((int*)W)[WS_VICTIDX]=vict; ((int*)W)[WS_OVERCAP]=over; W[WS_CAPF]=cap;
    svict=vict; sover=over;
  }
  __syncthreads();
  if(sover){
    int vict=svict;
    mem_out[(size_t)vict*ND + t] = 0.f;      // 512 threads
    if(t==0){
      at_out[vict] = -99999.f;
      float step=(float)stepp[0];
      float a=(step+99999.f)*1e-3f;
      float sum=0.f;
      for(int k=0;k<32;++k) sum += eg_w[k]*fmaxf(a*lru_w[k]+lru_b[k],0.f);
      for(int j=0;j<32;++j) sum += eg_w[32+j]*sigf(conf_b[j]); // zero row -> lin = bias
      erase_out[vict]=sigf(sum+eg_b[0]);
    }
  }
}

// ---------- K3b: argmax(erase_scores) ----------
__global__ void k3b_wargmax(const float* erase_out, float* W){
  int s=blockIdx.x*blockDim.x+threadIdx.x;
  ull key=(((ull)ordkey(erase_out[s]))<<32)|(ull)(0xFFFFFFFFu-(unsigned)s);
  for(int o=16;o>0;o>>=1){ ull oth=shfl_xor_u64(key,o); if(oth>key)key=oth; }
  if((threadIdx.x&31)==0) atomicMax((ull*)(W+WS_WPACK), key);
}

// ---------- K4: per-b raw max (sims) + top3 (sims2 = sims w/ victim col zeroed) ----------
__global__ void k4_top(float* W){
  __shared__ float lv[256*3]; __shared__ int li[256*3]; __shared__ float lmax[256];
  int b=blockIdx.x, t=threadIdx.x;
  const float* sims = W + WS_SIMS + (size_t)b*NS;
  int over=((int*)W)[WS_OVERCAP]; int vict=((int*)W)[WS_VICTIDX];
  float v0=-1e30f,v1=-1e30f,v2=-1e30f; int i0=0,i1=0,i2=0; float rmax=-1e30f;
  for(int s=t;s<NS;s+=256){
    float raw=sims[s];
    rmax=fmaxf(rmax,raw);
    float v=(over&&s==vict)?0.f:raw;
    if(v>v0){ v2=v1;i2=i1; v1=v0;i1=i0; v0=v;i0=s; }
    else if(v>v1){ v2=v1;i2=i1; v1=v;i1=s; }
    else if(v>v2){ v2=v;i2=s; }
  }
  lv[t*3]=v0; lv[t*3+1]=v1; lv[t*3+2]=v2;
  li[t*3]=i0; li[t*3+1]=i1; li[t*3+2]=i2;
  lmax[t]=rmax;
  __syncthreads();
  if(t==0){
    float bv0=-1e30f,bv1=-1e30f,bv2=-1e30f; int bi0=0,bi1=0,bi2=0;
    for(int k=0;k<256*3;++k){
      float v=lv[k]; int i=li[k];
      if((v>bv0)||(v==bv0&&i<bi0)){ bv2=bv1;bi2=bi1; bv1=bv0;bi1=bi0; bv0=v;bi0=i; }
      else if((v>bv1)||(v==bv1&&i<bi1)){ bv2=bv1;bi2=bi1; bv1=v;bi1=i; }
      else if((v>bv2)||(v==bv2&&i<bi2)){ bv2=v;bi2=i; }
    }
    W[WS_TOPV+b*3]=bv0; W[WS_TOPV+b*3+1]=bv1; W[WS_TOPV+b*3+2]=bv2;
    ((int*)W)[WS_TOPI+b*3]=bi0; ((int*)W)[WS_TOPI+b*3+1]=bi1; ((int*)W)[WS_TOPI+b*3+2]=bi2;
    float m=-1e30f; for(int k=0;k<256;++k) m=fmaxf(m,lmax[k]);
    W[WS_MAXSIM+b]=m;
  }
}

// ---------- K5a/b/c: store-gating MLP chain ----------
__global__ void k5a(const float* nc, const float* q, const float* rw1, const float* rb1,
                    const float* nw, const float* nb, float* W){
  int b=blockIdx.x, j=threadIdx.x;
  float acc=rb1[j];
  const float* w=rw1+(size_t)j*1024;
  for(int k=0;k<512;++k) acc+=nc[b*512+k]*w[k];
  for(int k=0;k<512;++k) acc+=q [b*512+k]*w[512+k];
  W[WS_H1+b*256+j]=acc;
  if(j<128){
    float a2=nb[j];
    const float* wn=nw+(size_t)j*512;
    for(int k=0;k<512;++k) a2+=nc[b*512+k]*wn[k];
    W[WS_NOVLIN+b*128+j]=a2;
  }
}
__global__ void k5b(const float* g, const float* bb, float* W){
  __shared__ float red[256];
  int b=blockIdx.x,t=threadIdx.x;
  float x=W[WS_H1+b*256+t];
  red[t]=x; __syncthreads();
  for(int o=128;o>0;o>>=1){ if(t<o) red[t]+=red[t+o]; __syncthreads(); }
  float m=red[0]/256.f; __syncthreads();
  float d=x-m; red[t]=d*d; __syncthreads();
  for(int o=128;o>0;o>>=1){ if(t<o) red[t]+=red[t+o]; __syncthreads(); }
  float var=red[0]/256.f;
  W[WS_HLN+b*256+t]=fmaxf(d*rsqrtf(var+1e-5f)*g[t]+bb[t],0.f);
}
__global__ void k5c(const float* rw2, const float* rb2, const float* gw, const float* gb, float* W){
  __shared__ float red[128];
  int b=blockIdx.x,j=threadIdx.x;
  float acc=rb2[j];
  const float* w=rw2+(size_t)j*256;
  for(int k=0;k<256;++k) acc+=W[WS_HLN+b*256+k]*w[k];
  float r=fmaxf(acc,0.f);
  float nf=sigf(W[WS_NOVLIN+b*128+j]);
  red[j]=(r+nf)*gw[j];
  __syncthreads();
  for(int o=64;o>0;o>>=1){ if(j<o) red[j]+=red[j+o]; __syncthreads(); }
  if(j==0) W[WS_SSC+b]=sigf(red[0]+gb[0]);
}

// ---------- K6a/b: conflict-detect / drift-strength MLPs over 96 pairs ----------
__global__ void k6a(const float* nc, const float* mem_out,
                    const float* dw1, const float* db1, const float* sw1, const float* sb1, float* W){
  int i=blockIdx.x, j=threadIdx.x;
  int b=i/3;
  int s=((int*)W)[WS_TOPI+i];
  const float* mrow = mem_out+(size_t)s*512;   // post-victim, pre-scatter
  W[WS_OLD+i*512+j]     = mrow[j];
  W[WS_OLD+i*512+256+j] = mrow[256+j];
  float ad=db1[j], as=sb1[j];
  const float* wd=dw1+(size_t)j*1024;
  const float* ws2=sw1+(size_t)j*1024;
  for(int k=0;k<512;++k){ float x=nc[b*512+k]; ad+=x*wd[k];     as+=x*ws2[k];     }
  for(int k=0;k<512;++k){ float x=mrow[k];     ad+=x*wd[512+k]; as+=x*ws2[512+k]; }
  W[WS_HD+i*256+j]=fmaxf(ad,0.f);
  W[WS_HS+i*256+j]=fmaxf(as,0.f);
}
__global__ void k6b(const float* dw2, const float* db2, const float* sw2, const float* sb2, float* W){
  __shared__ float rd[256], rs[256];
  int i=blockIdx.x,t=threadIdx.x;
  rd[t]=W[WS_HD+i*256+t]*dw2[t];
  rs[t]=W[WS_HS+i*256+t]*sw2[t];
  __syncthreads();
  for(int o=128;o>0;o>>=1){ if(t<o){rd[t]+=rd[t+o]; rs[t]+=rs[t+o];} __syncthreads(); }
  if(t==0){
    float prob=sigf(rd[0]+db2[0]);
    float st  =sigf(rs[0]+sb2[0]);
    float tv=W[WS_TOPV+i];
    int mask=(tv>0.7f)&&(tv<0.99f);
    W[WS_STR+i]=st;
    ((int*)W)[WS_ACT+i]=mask&&(prob>0.5f);
    if(mask) atomicAdd((unsigned*)W+WS_CONFCNT,1u);
  }
}

// ---------- K7a: winners (last-writer-wins), drifted_new[0], gate scalars, write_idx ----------
__global__ void k7a(const float* nc, float* W){
  __shared__ int sdw0;
  int t=threadIdx.x;
  int* Wi=(int*)W;
  if(t<96){
    int act=Wi[WS_ACT+t];
    int win=act;
    if(act){
      int s=Wi[WS_TOPI+t];
      for(int j=t+1;j<96;++j) if(Wi[WS_ACT+j] && Wi[WS_TOPI+j]==s){ win=0; break; }
    }
    Wi[WS_WIN+t]=win;
  }
  if(t==0){
    int d=-1;
    for(int i=0;i<3;++i) if(Wi[WS_ACT+i]) d=i;  // last active in b=0 group
    sdw0=d;
  }
  __syncthreads();
  int d=sdw0;
  for(int c=t;c<512;c+=256){
    float x=nc[c], dn=x;
    if(d>=0){ float st=W[WS_STR+d]; float od=W[WS_OLD+d*512+c]; dn=(1.f-st)*x+st*0.5f*(x+od); }
    W[WS_DRIFT0+c]=dn;
  }
  if(t==0){
    float sm=0.f,nm=0.f,rv=0.f;
    for(int b=0;b<32;++b){ sm+=W[WS_SSC+b]; float ms=W[WS_MAXSIM+b]; nm+=(1.f-ms)*0.5f; rv+=sigf(ms); }
    sm/=32.f; nm/=32.f; rv/=32.f;
    W[WS_STMEAN]=sm; W[WS_NOVMEAN]=nm; W[WS_RELEV]=rv;
    Wi[WS_SHOULD]=(sm>STORE_TH)&&(nm>MIN_NOV);
    int widx;
    if(Wi[WS_OVERCAP]) widx=Wi[WS_VICTIDX];
    else { ull key=*(ull*)(W+WS_WPACK); widx=(int)(0xFFFFFFFFu-(unsigned)(key&0xFFFFFFFFull)); }
    Wi[WS_WIDX]=widx;
  }
}

// ---------- K7b: apply cand_old scatter for winners ----------
__global__ void k7b(const float* nc, float* mem_out, float* W){
  int i=blockIdx.x,t=threadIdx.x;
  if(!((int*)W)[WS_WIN+i]) return;
  int b=i/3, s=((int*)W)[WS_TOPI+i];
  float st=W[WS_STR+i];
  for(int c=t;c<512;c+=256){
    float od=W[WS_OLD+i*512+c];
    float x=nc[b*512+c];
    mem_out[(size_t)s*512+c]=(1.f-st)*od+st*0.5f*(x+od);
  }
}

// ---------- K8: conditional store of drifted_new[0] ----------
__global__ void k8_store(float* mem_out, float* at_out, const int* stepp, float* W){
  if(!((int*)W)[WS_SHOULD]) return;
  int widx=((int*)W)[WS_WIDX], t=threadIdx.x;
  for(int c=t;c<512;c+=256) mem_out[(size_t)widx*512+c]=W[WS_DRIFT0+c];
  if(t==0) at_out[widx]=(float)stepp[0];
}

// ---------- K9: deduped list of modified rows ----------
__global__ void k9_list(float* W){
  if(threadIdx.x) return;
  int* Wi=(int*)W;
  int lst[100]; int n=0;
  lst[n++]=Wi[WS_VICTIDX];
  lst[n++]=Wi[WS_WIDX];
  for(int i=0;i<96;++i) if(Wi[WS_WIN+i]) lst[n++]=Wi[WS_TOPI+i];
  int m=0;
  for(int i=0;i<n;++i){
    int dup=0;
    for(int j=0;j<m;++j) if(Wi[WS_LIST+j]==lst[i]){dup=1;break;}
    if(!dup) Wi[WS_LIST+m++]=lst[i];
  }
  Wi[WS_LISTCNT]=m;
}

// ---------- K10: per-modified-row |diff| sums (deterministic) ----------
__global__ void k10_change(const float* mem0, const float* mem_out, float* W){
  __shared__ float red[256];
  int r=blockIdx.x, t=threadIdx.x;
  if(r>=((int*)W)[WS_LISTCNT]) return;
  int s=((int*)W)[WS_LIST+r];
  float p=0.f;
  for(int c=t;c<512;c+=256) p+=fabsf(mem_out[(size_t)s*512+c]-mem0[(size_t)s*512+c]);
  red[t]=p; __syncthreads();
  for(int o=128;o>0;o>>=1){ if(t<o) red[t]+=red[t+o]; __syncthreads(); }
  if(t==0) W[WS_ROWSUM+r]=red[0];
}

// ---------- K11: finalize scalars ----------
__global__ void k11_final(const float* decay_in, const float* ema_in, float* out_sc, float* W){
  if(threadIdx.x) return;
  int cnt=((int*)W)[WS_LISTCNT];
  float chsum=0.f;
  for(int r=0;r<cnt;++r) chsum+=W[WS_ROWSUM+r];
  float change=chsum/((float)NS*(float)ND);
  float ema=0.9f*ema_in[0]+0.1f*change;
  float adj=(ema>STAB)?-0.005f:0.005f;
  float decay=fminf(fmaxf(decay_in[0]+adj,0.01f),0.2f);
  out_sc[0]=W[WS_STMEAN];
  out_sc[1]=W[WS_NOVMEAN];
  out_sc[2]=W[WS_RELEV];
  out_sc[3]=W[WS_CAPF];
  out_sc[4]=(float)(((unsigned*)W)[WS_CONFCNT]);
  out_sc[5]=decay;
  out_sc[6]=ema;
}

extern "C" void kernel_launch(void* const* d_in, const int* in_sizes, int n_in,
                              void* d_out, int out_size, void* d_ws, size_t ws_size,
                              hipStream_t stream){
  const float* nc  =(const float*)d_in[0];
  const float* q   =(const float*)d_in[1];
  const float* mem0=(const float*)d_in[2];
  const float* at  =(const float*)d_in[3];
  const float* rw1 =(const float*)d_in[4];
  const float* rb1 =(const float*)d_in[5];
  const float* rg  =(const float*)d_in[6];
  const float* rbb =(const float*)d_in[7];
  const float* rw2 =(const float*)d_in[8];
  const float* rb2 =(const float*)d_in[9];
  const float* nw  =(const float*)d_in[10];
  const float* nb  =(const float*)d_in[11];
  const float* gw  =(const float*)d_in[12];
  const float* gb  =(const float*)d_in[13];
  const float* lruw=(const float*)d_in[14];
  const float* lrub=(const float*)d_in[15];
  const float* cw  =(const float*)d_in[16];
  const float* cb  =(const float*)d_in[17];
  const float* egw =(const float*)d_in[18];
  const float* egb =(const float*)d_in[19];
  const float* dw1 =(const float*)d_in[20];
  const float* db1 =(const float*)d_in[21];
  const float* dw2 =(const float*)d_in[22];
  const float* db2 =(const float*)d_in[23];
  const float* sw1 =(const float*)d_in[24];
  const float* sb1 =(const float*)d_in[25];
  const float* sw2 =(const float*)d_in[26];
  const float* sb2 =(const float*)d_in[27];
  const float* decay=(const float*)d_in[28];
  const float* ema  =(const float*)d_in[29];
  const int*   stepp=(const int*)d_in[30];

  float* out=(float*)d_out;
  float* mem_out=out+OUT_MEM;
  float* at_out =out+OUT_AT;
  float* er_out =out+OUT_ER;
  float* sc_out =out+OUT_SC;
  float* W=(float*)d_ws;

  k0_prep   <<<32,256,0,stream>>>(nc,cw,W);
  k1_agemax <<<NS/256,256,0,stream>>>(at,stepp,W);
  k2_main   <<<NS/128,256,(size_t)(8*16*ND*2),stream>>>(mem0,at,mem_out,at_out,er_out,W,
                                                        cb,egw,egb,lruw,lrub,stepp);
  k3_fix    <<<1,512,0,stream>>>(mem_out,at_out,er_out,W,cb,egw,egb,lruw,lrub,stepp);
  k3b_wargmax<<<NS/256,256,0,stream>>>(er_out,W);
  k4_top    <<<32,256,0,stream>>>(W);
  k5a       <<<32,256,0,stream>>>(nc,q,rw1,rb1,nw,nb,W);
  k5b       <<<32,256,0,stream>>>(rg,rbb,W);
  k5c       <<<32,128,0,stream>>>(rw2,rb2,gw,gb,W);
  k6a       <<<96,256,0,stream>>>(nc,mem_out,dw1,db1,sw1,sb1,W);
  k6b       <<<96,256,0,stream>>>(dw2,db2,sw2,sb2,W);
  k7a       <<<1,256,0,stream>>>(nc,W);
  k7b       <<<96,256,0,stream>>>(nc,mem_out,W);
  k8_store  <<<1,256,0,stream>>>(mem_out,at_out,stepp,W);
  k9_list   <<<1,64,0,stream>>>(W);
  k10_change<<<98,256,0,stream>>>(mem0,mem_out,W);
  k11_final <<<1,1,0,stream>>>(decay,ema,sc_out,W);
}
